// GatedTinyCrossAttnAdapter_23313082482836
// MI455X (gfx1250) — matrix operand order repaired and checked by reading, not compiled, not run on hardware
//
#include <hip/hip_runtime.h>

typedef __attribute__((ext_vector_type(16))) _Float16 v16h;
typedef __attribute__((ext_vector_type(8)))  _Float16 v8h;
typedef __attribute__((ext_vector_type(8)))  float    v8f;

constexpr int cH   = 4096;   // hidden
constexpr int cD   = 1024;   // mem dim
constexpr int cA   = 512;    // attn dim
constexpr int cB   = 4;
constexpr int cLQ  = 2048;
constexpr int cTM  = 1500;
constexpr int cTMP = 1504;   // TM padded to 32
constexpr float cEPS = 1e-5f;
constexpr float cSCALE = 0.04419417382415922f; // 1/sqrt(512)
constexpr float cNEG  = -60000.0f;             // finite in f16; acts as -inf

union U32H2 { unsigned int u; _Float16 h[2]; };

__device__ __forceinline__ v16h cat16(v8h lo, v8h hi) {
  return __builtin_shufflevector(lo, hi, 0, 1, 2, 3, 4, 5, 6, 7,
                                 8, 9, 10, 11, 12, 13, 14, 15);
}
// A-matrix 16x32 f16 frag: halfwords [8*half,+8) and [16+8*half,+8).
__device__ __forceinline__ v16h load_a_frag(const _Float16* p, int half) {
  return cat16(*(const v8h*)(p + 8 * half), *(const v8h*)(p + 16 + 8 * half));
}
// B-matrix 32x16 f16 frag: halfwords [8*half,+8) and [16+8*half,+8).
__device__ __forceinline__ v16h load_b_frag(const _Float16* p, int half) {
  return cat16(*(const v8h*)(p + 8 * half), *(const v8h*)(p + 16 + 8 * half));
}

// ---------------------------------------------------------------- reductions
__device__ __forceinline__ void block_reduce2(float& s, float& s2, float* red) {
  #pragma unroll
  for (int off = 16; off >= 1; off >>= 1) {
    s  += __shfl_xor(s,  off, 32);
    s2 += __shfl_xor(s2, off, 32);
  }
  int wv = (int)(threadIdx.x >> 5);
  if ((threadIdx.x & 31u) == 0u) { red[wv] = s; red[8 + wv] = s2; }
  __syncthreads();
  float ts = 0.f, ts2 = 0.f;
  #pragma unroll
  for (int i = 0; i < 8; ++i) { ts += red[i]; ts2 += red[8 + i]; }
  s = ts; s2 = ts2;
  __syncthreads();
}

// ------------------------------------------------------------ LN(in) -> f16
__global__ __launch_bounds__(256)
void ln_in_kernel(const float* __restrict__ x, const float* __restrict__ w,
                  const float* __restrict__ b, _Float16* __restrict__ y) {
  __shared__ float row[cH];
  __shared__ float red[16];
  size_t r = blockIdx.x;
  const float* xp = x + r * cH;
  float s = 0.f, s2 = 0.f;
  for (int h = threadIdx.x; h < cH; h += 256) {
    float v = xp[h];
    row[h] = v;
    s += v; s2 += v * v;
  }
  block_reduce2(s, s2, red);
  float mu  = s * (1.0f / cH);
  float var = s2 * (1.0f / cH) - mu * mu;
  float rs  = rsqrtf(var + cEPS);
  for (int h = threadIdx.x; h < cH; h += 256)
    y[r * cH + h] = (_Float16)((row[h] - mu) * rs * w[h] + b[h]);
}

// --------------------------------------------------------------- f32 -> f16
__global__ __launch_bounds__(256)
void f32_to_f16_kernel(const float* __restrict__ src, _Float16* __restrict__ dst, long n) {
  long i = (long)blockIdx.x * 256 + threadIdx.x;
  long stride = (long)gridDim.x * 256;
  for (; i < n; i += stride) dst[i] = (_Float16)src[i];
}

// ------------------------- V transpose: V[b][t][a] -> Vt[b][a][t] (padded 0)
__global__ __launch_bounds__(256)
void transpose_v_kernel(const _Float16* __restrict__ V, _Float16* __restrict__ Vt) {
  __shared__ _Float16 tile[32][33];
  int b  = blockIdx.z;
  int t0 = blockIdx.x * 32, a0 = blockIdx.y * 32;
  int tx = threadIdx.x & 31, ty = threadIdx.x >> 5;   // 32 x 8
  for (int i = ty; i < 32; i += 8) {
    int t = t0 + i;
    _Float16 v = (_Float16)0.f;
    if (t < cTM) v = V[(size_t)(b * cTM + t) * cA + a0 + tx];
    tile[i][tx] = v;
  }
  __syncthreads();
  for (int i = ty; i < 32; i += 8) {
    int a = a0 + i;
    Vt[(size_t)(b * cA + a) * cTMP + t0 + tx] = tile[tx][i];
  }
}

// -------------------------------------------------- generic WMMA tiled GEMM
// C[M,N] = A[M,K](f16,row) * B[K,N](f16,row) + bias ; block tile 64x128, BK=32
// Double-buffered LDS; A tile staged with CDNA5 async-to-LDS DMA (ASYNCcnt),
// B tile prefetched to VGPRs during WMMA and stored transposed after.
template<typename OutT>
__global__ __launch_bounds__(256)
void gemm_wmma(const _Float16* __restrict__ Ag, const _Float16* __restrict__ Bg,
               OutT* __restrict__ Cg, const float* __restrict__ bias,
               int M, int N, int K) {
  __shared__ __align__(16) _Float16 As[2][64][32];   // (m,k)
  __shared__ __align__(16) _Float16 Bs[2][128][32];  // transposed: (n,k)
  const int tid  = threadIdx.x;
  const int wave = tid >> 5, lane = tid & 31;
  const int wy = wave >> 2, wx = wave & 3;           // 2 x 4 wave grid
  const int half = lane >> 4, l16 = lane & 15;
  const int bm0 = blockIdx.y * 64, bn0 = blockIdx.x * 128;

  // A staging: one async b128 per thread (64 rows x 4 chunks)
  const int am = tid >> 2;
  const int ac = (tid & 3) << 3;                     // halfword chunk base
  int agm = bm0 + am; agm = (agm < M) ? agm : (M - 1);
  const _Float16* arow = Ag + (size_t)agm * K + ac;

  auto issue_async_A = [&](int buf, int k0) {
    unsigned lds = (unsigned)(unsigned long long)(void*)&As[buf][am][ac];
    unsigned long long gp = (unsigned long long)(arow + k0);
    asm volatile("global_load_async_to_lds_b128 %0, %1, off"
                 :: "v"(lds), "v"(gp) : "memory");
  };
  auto load_B = [&](int k0, unsigned (&breg)[8]) {
    #pragma unroll
    for (int it = 0; it < 8; ++it) {
      int slot = tid + it * 256;
      int kk = slot >> 6;
      int np = (slot & 63) << 1;
      int gn = bn0 + np; gn = (gn + 1 < N) ? gn : (N - 2);
      breg[it] = *(const unsigned int*)(Bg + (size_t)(k0 + kk) * N + gn);
    }
  };
  auto store_B = [&](int buf, const unsigned (&breg)[8]) {
    #pragma unroll
    for (int it = 0; it < 8; ++it) {
      int slot = tid + it * 256;
      int kk = slot >> 6;
      int np = (slot & 63) << 1;
      U32H2 u; u.u = breg[it];
      Bs[buf][np][kk]     = u.h[0];
      Bs[buf][np + 1][kk] = u.h[1];
    }
  };

  // prologue: stage tile 0
  {
    unsigned breg[8];
    load_B(0, breg);
    issue_async_A(0, 0);
    store_B(0, breg);
    asm volatile("s_wait_asynccnt 0x0" ::: "memory");
    __syncthreads();
  }

  v8f acc[2][2] = {};
  int cur = 0;
  for (int k0 = 0; k0 < K; k0 += 32) {
    const int nxt = cur ^ 1;
    const bool pf = (k0 + 32) < K;
    unsigned breg[8];
    if (pf) {                       // issue next-tile traffic before compute
      load_B(k0 + 32, breg);
      issue_async_A(nxt, k0 + 32);
    }

    v16h afrag[2], bfrag[2];
    #pragma unroll
    for (int ms = 0; ms < 2; ++ms)
      afrag[ms] = load_a_frag(&As[cur][wy * 32 + ms * 16 + l16][0], half);
    #pragma unroll
    for (int ns = 0; ns < 2; ++ns)
      bfrag[ns] = load_b_frag(&Bs[cur][wx * 32 + ns * 16 + l16][0], half);
    #pragma unroll
    for (int ms = 0; ms < 2; ++ms)
      #pragma unroll
      for (int ns = 0; ns < 2; ++ns)
        acc[ms][ns] = __builtin_amdgcn_wmma_f32_16x16x32_f16(
            false, afrag[ms], false, bfrag[ns], (short)0, acc[ms][ns], false, false);

    if (pf) store_B(nxt, breg);
    asm volatile("s_wait_asynccnt 0x0" ::: "memory");
    __syncthreads();
    cur = nxt;
  }

  #pragma unroll
  for (int ms = 0; ms < 2; ++ms)
    #pragma unroll
    for (int ns = 0; ns < 2; ++ns) {
      int n = bn0 + wx * 32 + ns * 16 + l16;
      if (n >= N) continue;
      float bvv = bias ? bias[n] : 0.f;
      #pragma unroll
      for (int r = 0; r < 8; ++r) {
        int m = bm0 + wy * 32 + ms * 16 + r + 8 * half;
        if (m < M) Cg[(size_t)m * N + n] = (OutT)(acc[ms][ns][r] + bvv);
      }
    }
}

// ----------------------------------------------------------- attention core
// one block = one batch b, 32 query rows. Score strip kept in f16 and
// softmaxed in place: 94 KB LDS -> 3 blocks/WGP for latency hiding.
// Each wave runs TWO adjacent column tiles per step (shared A-frag, two
// independent B streams, two accumulators) so loads overlap WMMA.
__global__ __launch_bounds__(256)
void attn_kernel(const _Float16* __restrict__ Qg, const _Float16* __restrict__ Kg,
                 const _Float16* __restrict__ Vtg, const unsigned char* __restrict__ maskg,
                 _Float16* __restrict__ Ctx) {
  extern __shared__ char smem_raw[];
  _Float16* Ssh = (_Float16*)smem_raw;     // 32 x cTMP f16: scores, then probs

  const int tid  = threadIdx.x;
  const int wave = tid >> 5, lane = tid & 31;
  const int half = lane >> 4, l16 = lane & 15;
  const int bb = blockIdx.y;
  const int q0 = blockIdx.x * 32;
  const int ms = wave & 1;                 // q sub-tile fixed per wave
  const int wg = wave >> 1;                // 0..3 column-tile group

  // ---- Phase 1: S = scale * Q Kt ; two K streams per wave
  const int NT = cTMP / 16;                // 94 column tiles (even)
  {
    const _Float16* qrow = Qg + (size_t)(bb * cLQ + q0 + ms * 16 + l16) * cA;
    for (int nt = wg * 2; nt < NT; nt += 8) {
      int t0 = nt * 16 + l16;
      int t1 = t0 + 16;
      int tc0 = (t0 < cTM) ? t0 : (cTM - 1);
      int tc1 = (t1 < cTM) ? t1 : (cTM - 1);
      const _Float16* k0r = Kg + (size_t)(bb * cTM + tc0) * cA;
      const _Float16* k1r = Kg + (size_t)(bb * cTM + tc1) * cA;
      v8f acc0 = {}, acc1 = {};
      #pragma unroll
      for (int ka = 0; ka < cA; ka += 32) {
        v16h af = load_a_frag(qrow + ka, half);
        v16h b0 = load_b_frag(k0r + ka, half);
        v16h b1 = load_b_frag(k1r + ka, half);
        acc0 = __builtin_amdgcn_wmma_f32_16x16x32_f16(false, af, false, b0,
                                                      (short)0, acc0, false, false);
        acc1 = __builtin_amdgcn_wmma_f32_16x16x32_f16(false, af, false, b1,
                                                      (short)0, acc1, false, false);
      }
      bool m0 = (t0 >= cTM) || (maskg[bb * cTM + tc0] != 0);
      bool m1 = (t1 >= cTM) || (maskg[bb * cTM + tc1] != 0);
      #pragma unroll
      for (int r = 0; r < 8; ++r) {
        int ml = ms * 16 + r + 8 * half;
        Ssh[(size_t)ml * cTMP + t0] = (_Float16)(m0 ? cNEG : acc0[r] * cSCALE);
        Ssh[(size_t)ml * cTMP + t1] = (_Float16)(m1 ? cNEG : acc1[r] * cSCALE);
      }
    }
  }
  __syncthreads();

  // ---- Phase 2: in-place row softmax (8 lanes per row, shuffle reductions)
  {
    int m   = tid >> 3;
    int sub = tid & 7;
    _Float16* srow = Ssh + (size_t)m * cTMP;
    float mx = cNEG;
    for (int t = sub; t < cTM; t += 8) mx = fmaxf(mx, (float)srow[t]);
    #pragma unroll
    for (int off = 4; off >= 1; off >>= 1) mx = fmaxf(mx, __shfl_xor(mx, off, 32));
    float sum = 0.f;
    for (int t = sub; t < cTM; t += 8) {
      float e = __expf((float)srow[t] - mx);
      sum += e;
      srow[t] = (_Float16)e;
    }
    #pragma unroll
    for (int off = 4; off >= 1; off >>= 1) sum += __shfl_xor(sum, off, 32);
    float inv = 1.0f / sum;
    for (int t = sub; t < cTM; t += 8) srow[t] = (_Float16)((float)srow[t] * inv);
    for (int t = cTM + sub; t < cTMP; t += 8) srow[t] = (_Float16)0.f;
  }
  __syncthreads();

  // ---- Phase 3: context = P Vt ; two Vt streams per wave
  {
    const _Float16* prow = Ssh + (size_t)(ms * 16 + l16) * cTMP;
    for (int na = wg * 2; na < cA / 16; na += 8) {
      int a0 = na * 16 + l16;
      int a1 = a0 + 16;
      const _Float16* v0r = Vtg + (size_t)(bb * cA + a0) * cTMP;
      const _Float16* v1r = Vtg + (size_t)(bb * cA + a1) * cTMP;
      v8f acc0 = {}, acc1 = {};
      for (int kt = 0; kt < cTMP; kt += 32) {
        v16h af = load_a_frag(prow + kt, half);
        v16h b0 = load_b_frag(v0r + kt, half);
        v16h b1 = load_b_frag(v1r + kt, half);
        acc0 = __builtin_amdgcn_wmma_f32_16x16x32_f16(false, af, false, b0,
                                                      (short)0, acc0, false, false);
        acc1 = __builtin_amdgcn_wmma_f32_16x16x32_f16(false, af, false, b1,
                                                      (short)0, acc1, false, false);
      }
      #pragma unroll
      for (int r = 0; r < 8; ++r) {
        int m = q0 + ms * 16 + r + 8 * half;
        Ctx[(size_t)(bb * cLQ + m) * cA + a0] = (_Float16)acc0[r];
        Ctx[(size_t)(bb * cLQ + m) * cA + a1] = (_Float16)acc1[r];
      }
    }
  }
}

// -------------------------------------- residual + LN(out) + sigmoid gating
__global__ __launch_bounds__(256)
void final_kernel(const float* __restrict__ hidden, const float* __restrict__ ch,
                  const float* __restrict__ w, const float* __restrict__ b,
                  const float* __restrict__ gate_logit, float* __restrict__ out) {
  __shared__ float row[cH];
  __shared__ float red[16];
  size_t r = blockIdx.x;
  const float* hp = hidden + r * cH;
  const float* cp = ch + r * cH;
  float s = 0.f, s2 = 0.f;
  for (int h = threadIdx.x; h < cH; h += 256) {
    float v = hp[h] + cp[h];            // bo already folded into ch by the GEMM
    row[h] = v;
    s += v; s2 += v * v;
  }
  block_reduce2(s, s2, red);
  float mu  = s * (1.0f / cH);
  float var = s2 * (1.0f / cH) - mu * mu;
  float rs  = rsqrtf(var + cEPS);
  float g   = 1.0f / (1.0f + __expf(-gate_logit[0]));
  for (int h = threadIdx.x; h < cH; h += 256) {
    float hv  = hp[h];
    float lnv = (row[h] - mu) * rs * w[h] + b[h];
    out[r * cH + h] = hv + g * (lnv - hv);
  }
}

// ---------------------------------------------------------------- dispatch
extern "C" void kernel_launch(void* const* d_in, const int* in_sizes, int n_in,
                              void* d_out, int out_size, void* d_ws, size_t ws_size,
                              hipStream_t stream) {
  (void)in_sizes; (void)n_in; (void)out_size; (void)ws_size;
  const float* hidden  = (const float*)d_in[0];
  const float* mem     = (const float*)d_in[1];
  const unsigned char* mask = (const unsigned char*)d_in[2];
  const float* Wq = (const float*)d_in[3];
  const float* bq = (const float*)d_in[4];
  const float* Wk = (const float*)d_in[5];
  const float* bk = (const float*)d_in[6];
  const float* Wv = (const float*)d_in[7];
  const float* bv = (const float*)d_in[8];
  const float* Wo = (const float*)d_in[9];
  const float* bo = (const float*)d_in[10];
  const float* ln_in_w  = (const float*)d_in[11];
  const float* ln_in_b  = (const float*)d_in[12];
  const float* ln_out_w = (const float*)d_in[13];
  const float* ln_out_b = (const float*)d_in[14];
  const float* gate_logit = (const float*)d_in[15];
  float* out = (float*)d_out;

  char* ws = (char*)d_ws;
  size_t off = 0;
  auto take = [&](size_t bytes) -> char* {
    char* p = ws + off;
    off = (off + bytes + 255) & ~(size_t)255;
    return p;
  };
  _Float16* h_norm = (_Float16*)take((size_t)cB * cLQ * cH * 2);
  _Float16* mem16  = (_Float16*)take((size_t)cB * cTM * cD * 2);
  _Float16* Wq16   = (_Float16*)take((size_t)cH * cA * 2);
  _Float16* Wk16   = (_Float16*)take((size_t)cD * cA * 2);
  _Float16* Wv16   = (_Float16*)take((size_t)cD * cA * 2);
  _Float16* Wo16   = (_Float16*)take((size_t)cA * cH * 2);
  _Float16* Q16    = (_Float16*)take((size_t)cB * cLQ * cA * 2);
  _Float16* K16    = (_Float16*)take((size_t)cB * cTM * cA * 2);
  _Float16* V16    = (_Float16*)take((size_t)cB * cTM * cA * 2);
  _Float16* Vt16   = (_Float16*)take((size_t)cB * cA * cTMP * 2);
  _Float16* C16    = (_Float16*)take((size_t)cB * cLQ * cA * 2);
  float*    CH32   = (float*)   take((size_t)cB * cLQ * cH * 4);

  // 1) LN-in (fp32 stats) -> f16 activations; weight/mem down-converts
  ln_in_kernel<<<cB * cLQ, 256, 0, stream>>>(hidden, ln_in_w, ln_in_b, h_norm);
  f32_to_f16_kernel<<<1024, 256, 0, stream>>>(mem, mem16, (long)cB * cTM * cD);
  f32_to_f16_kernel<<<512,  256, 0, stream>>>(Wq, Wq16, (long)cH * cA);
  f32_to_f16_kernel<<<256,  256, 0, stream>>>(Wk, Wk16, (long)cD * cA);
  f32_to_f16_kernel<<<256,  256, 0, stream>>>(Wv, Wv16, (long)cD * cA);
  f32_to_f16_kernel<<<512,  256, 0, stream>>>(Wo, Wo16, (long)cA * cH);

  // 2) projections (WMMA, async-LDS double-buffered)
  gemm_wmma<_Float16><<<dim3(cA / 128, (cB * cLQ) / 64), 256, 0, stream>>>(
      h_norm, Wq16, Q16, bq, cB * cLQ, cA, cH);
  gemm_wmma<_Float16><<<dim3(cA / 128, (cB * cTM + 63) / 64), 256, 0, stream>>>(
      mem16, Wk16, K16, bk, cB * cTM, cA, cD);
  gemm_wmma<_Float16><<<dim3(cA / 128, (cB * cTM + 63) / 64), 256, 0, stream>>>(
      mem16, Wv16, V16, bv, cB * cTM, cA, cD);

  // 2b) transpose V for contiguous B-fragments in P@V
  transpose_v_kernel<<<dim3(cTMP / 32, cA / 32, cB), 256, 0, stream>>>(V16, Vt16);

  // 3) attention (WMMA; f16 score strip, in-place softmax, 94 KB LDS)
  size_t smem = (size_t)32 * cTMP * sizeof(_Float16);
  hipFuncSetAttribute(reinterpret_cast<const void*>(attn_kernel),
                      hipFuncAttributeMaxDynamicSharedMemorySize, (int)smem);
  attn_kernel<<<dim3(cLQ / 32, cB), 256, smem, stream>>>(Q16, K16, Vt16, mask, C16);

  // 4) output projection (WMMA, f32 out, bo folded in)
  gemm_wmma<float><<<dim3(cH / 128, (cB * cLQ) / 64), 256, 0, stream>>>(
      C16, Wo16, CH32, bo, cB * cLQ, cH, cA);

  // 5) residual + LN-out + gate
  final_kernel<<<cB * cLQ, 256, 0, stream>>>(hidden, CH32, ln_out_w, ln_out_b,
                                             gate_logit, out);
}
